// MultiHeadSelfAttention_72164040507732
// MI455X (gfx1250) — matrix operand order repaired and checked by reading, not compiled, not run on hardware
//
#include <hip/hip_runtime.h>
#include <hip/hip_bf16.h>
#include <stdint.h>

// MI455X / gfx1250, wave32. D = A*B + C via v_wmma_f32_16x16x32_f16.
// f16 storage, fp32 accumulation. Flash-attention with S^T trick (softmax
// per-lane, P^T lands in WMMA B-layout) + LDS-staged K/V chunks shared by
// all 8 waves of a workgroup (double buffered).

typedef _Float16 half_t;
typedef __attribute__((ext_vector_type(16))) _Float16 v16h;
typedef __attribute__((ext_vector_type(8)))  float    v8f;

union H16 { v16h v; uint4 q[2]; half_t e[16]; };

__device__ __forceinline__ v8f wmma16(v16h a, v16h b, v8f c) {
  // 8 args: (neg_a, A, neg_b, B, c_mod, C, reuse_a, reuse_b)
  return __builtin_amdgcn_wmma_f32_16x16x32_f16(false, a, false, b, (short)0, c, false, false);
}

// A-matrix tile (16 rows x 32 K), row-major source, leading dim ld (elements).
// Layout: lanes 0-15 hold K=0..15; lanes 16-31 hold K=16..31.
__device__ __forceinline__ v16h load_A(const half_t* base, int row, int ld, int k, int hi) {
  const half_t* p = base + (size_t)row * ld + k + hi * 16;
  H16 r;
  r.q[0] = *(const uint4*)(p);
  r.q[1] = *(const uint4*)(p + 8);
  return r.v;
}

// B-matrix tile (32 K x 16 cols); source stores each column contiguously (ld elems).
// Layout: lanes 0-15 hold K=0..15; lanes 16-31 hold K=16..31.
__device__ __forceinline__ v16h load_B(const half_t* base, int col, int ld, int k, int hi) {
  const half_t* p = base + (size_t)col * ld + k + hi * 16;
  H16 r;
  r.q[0] = *(const uint4*)(p);
  r.q[1] = *(const uint4*)(p + 8);
  return r.v;
}

// ---------------- fp32 -> f16 convert (vector x) ----------------
__global__ __launch_bounds__(256) void k_cvt(const float* __restrict__ in, half_t* __restrict__ out, int n) {
  int i = (blockIdx.x * 256 + threadIdx.x) * 4;
  if (i >= n) return;
  float4 f = *(const float4*)(in + i);
  union { half_t h[4]; uint2 u; } r;
  r.h[0] = (half_t)f.x; r.h[1] = (half_t)f.y; r.h[2] = (half_t)f.z; r.h[3] = (half_t)f.w;
  *(uint2*)(out + i) = r.u;
}

// ---------------- fp32 [R,C] -> f16 transposed [C,R] ----------------
__global__ __launch_bounds__(256) void k_transpose(const float* __restrict__ in, half_t* __restrict__ out,
                                                   int R, int C) {
  __shared__ float tile[32][33];
  int c0 = blockIdx.x * 32, r0 = blockIdx.y * 32;
  int tx = threadIdx.x & 31, ty = threadIdx.x >> 5;
#pragma unroll
  for (int i = 0; i < 32; i += 8)
    tile[ty + i][tx] = in[(size_t)(r0 + ty + i) * C + (c0 + tx)];
  __syncthreads();
#pragma unroll
  for (int i = 0; i < 32; i += 8)
    out[(size_t)(c0 + ty + i) * R + (r0 + tx)] = (half_t)tile[tx][ty + i];
}

// ---------------- QKV GEMM: [8192,1024] x [1024,3072] + bias ----------------
// Writes Q (scaled by 1/8) and K as [B,H,N,64] f16, V transposed as [B,H,64,N] f16.
__global__ __launch_bounds__(256) void k_gemm_qkv(const half_t* __restrict__ X, const half_t* __restrict__ Wt,
                                                  const float* __restrict__ bias,
                                                  half_t* __restrict__ Qo, half_t* __restrict__ Ko,
                                                  half_t* __restrict__ Vto) {
  const int lane = threadIdx.x & 31, wave = threadIdx.x >> 5;
  const int laneLo = lane & 15, hi = lane >> 4;
  const int Mbase = blockIdx.x * 128 + (wave & 1) * 64;
  const int Nbase = blockIdx.y * 128 + (wave >> 1) * 32;

  v8f acc[4][2] = {};
  for (int k = 0; k < 1024; k += 32) {
    v16h a0 = load_A(X, Mbase +  0 + laneLo, 1024, k, hi);
    v16h a1 = load_A(X, Mbase + 16 + laneLo, 1024, k, hi);
    v16h a2 = load_A(X, Mbase + 32 + laneLo, 1024, k, hi);
    v16h a3 = load_A(X, Mbase + 48 + laneLo, 1024, k, hi);
    v16h b0 = load_B(Wt, Nbase +  0 + laneLo, 1024, k, hi);
    v16h b1 = load_B(Wt, Nbase + 16 + laneLo, 1024, k, hi);
    acc[0][0] = wmma16(a0, b0, acc[0][0]);  acc[0][1] = wmma16(a0, b1, acc[0][1]);
    acc[1][0] = wmma16(a1, b0, acc[1][0]);  acc[1][1] = wmma16(a1, b1, acc[1][1]);
    acc[2][0] = wmma16(a2, b0, acc[2][0]);  acc[2][1] = wmma16(a2, b1, acc[2][1]);
    acc[3][0] = wmma16(a3, b0, acc[3][0]);  acc[3][1] = wmma16(a3, b1, acc[3][1]);
  }

#pragma unroll
  for (int nt = 0; nt < 2; nt++) {
    int c = Nbase + nt * 16 + laneLo;      // uniform 'which'/'h' across the tile
    float bv = bias[c];
    int which = c >> 10;                   // 0=q, 1=k, 2=v
    int cc = c & 1023;
    int h = cc >> 6, d = cc & 63;
#pragma unroll
    for (int mt = 0; mt < 4; mt++) {
      int row0 = Mbase + mt * 16 + hi * 8; // 8 consecutive rows per lane
      int b_ = row0 >> 11, n0 = row0 & 2047;
      if (which == 2) {
        union { half_t h8[8]; uint4 u; } r;
#pragma unroll
        for (int r8 = 0; r8 < 8; r8++) r.h8[r8] = (half_t)(acc[mt][nt][r8] + bv);
        *(uint4*)(Vto + ((size_t)(b_ * 16 + h) * 64 + d) * 2048 + n0) = r.u;
      } else {
        half_t* dst = (which == 0) ? Qo : Ko;
        float scl = (which == 0) ? 0.125f : 1.0f;   // SCALE = 64^-0.5 folded into Q (exact pow2)
        size_t base = ((size_t)(b_ * 16 + h) * 2048 + n0) * 64 + d;
#pragma unroll
        for (int r8 = 0; r8 < 8; r8++)
          dst[base + (size_t)r8 * 64] = (half_t)((acc[mt][nt][r8] + bv) * scl);
      }
    }
  }
}

// ---------------- Flash attention ----------------
// One workgroup = 8 waves, one (b,h), 256 queries (32 per wave). K/V chunks
// (32 keys) staged in double-buffered LDS shared by all waves; per-chunk WMMA
// A-tiles come from LDS (ds_load_b128), next chunk's global loads overlap.
__global__ __launch_bounds__(256) void k_attn(const half_t* __restrict__ Q, const half_t* __restrict__ K,
                                              const half_t* __restrict__ Vt, half_t* __restrict__ O) {
  __shared__ half_t Kbuf[2][32 * 64];   // [key within chunk][d]
  __shared__ half_t Vbuf[2][64 * 32];   // [d][key within chunk]

  const int tid = threadIdx.x;
  const int lane = tid & 31, wave = tid >> 5;
  const int laneLo = lane & 15, hi = lane >> 4;
  const int bh = blockIdx.x;                         // b*16 + h
  const int qbase = blockIdx.y * 256 + wave * 32;    // 32 queries per wave
  const half_t* Qp = Q  + (size_t)bh * 2048 * 64;
  const half_t* Kp = K  + (size_t)bh * 2048 * 64;
  const half_t* Vp = Vt + (size_t)bh * 64 * 2048;

  // Q^T B-operands: [qtile][dstep]; Q already carries the 1/8 scale.
  v16h bq[2][2];
#pragma unroll
  for (int qt = 0; qt < 2; qt++) {
    bq[qt][0] = load_B(Qp, qbase + qt * 16 + laneLo, 64,  0, hi);
    bq[qt][1] = load_B(Qp, qbase + qt * 16 + laneLo, 64, 32, hi);
  }

  v8f acc[2][4] = {};                                // [qtile][dtile] O^T accumulators
  float m_run[2] = {-3.4e38f, -3.4e38f};
  float l_run[2] = {0.f, 0.f};

  // Cooperative staging addressing (256 threads move 4KB K + 4KB V per chunk).
  const int krow = tid >> 3, kcol = (tid & 7) * 8;   // K chunk: 32 rows x 64 d
  const int vrow = tid >> 2, vcol = (tid & 3) * 8;   // V chunk: 64 rows x 32 keys

  // Prefetch chunk 0 into buffer 0.
  uint4 kreg = *(const uint4*)(Kp + (size_t)krow * 64 + kcol);
  uint4 vreg = *(const uint4*)(Vp + (size_t)vrow * 2048 + vcol);
  *(uint4*)(&Kbuf[0][krow * 64 + kcol]) = kreg;
  *(uint4*)(&Vbuf[0][vrow * 32 + vcol]) = vreg;

  int p = 0;
  for (int j = 0; j < 2048; j += 32, p ^= 1) {
    __syncthreads();                                 // buf[p] ready for all waves
    const bool more = (j + 32) < 2048;
    if (more) {                                      // overlap next chunk's global loads
      kreg = *(const uint4*)(Kp + (size_t)(j + 32 + krow) * 64 + kcol);
      vreg = *(const uint4*)(Vp + (size_t)vrow * 2048 + (j + 32) + vcol);
    }
    const half_t* kb = Kbuf[p];
    const half_t* vb = Vbuf[p];

    // K A-tiles (16 keys x 32 d) from LDS, shared across both q-tiles.
    v16h ka0 = load_A(kb,  0 + laneLo, 64,  0, hi);
    v16h ka1 = load_A(kb,  0 + laneLo, 64, 32, hi);
    v16h kc0 = load_A(kb, 16 + laneLo, 64,  0, hi);
    v16h kc1 = load_A(kb, 16 + laneLo, 64, 32, hi);

    H16 pb[2];
#pragma unroll
    for (int qt = 0; qt < 2; qt++) {
      // S^T tiles: T0 = keys j..j+15, T1 = keys j+16..j+31, queries as columns.
      v8f t0 = {}, t1 = {};
      t0 = wmma16(ka0, bq[qt][0], t0);  t0 = wmma16(ka1, bq[qt][1], t0);
      t1 = wmma16(kc0, bq[qt][0], t1);  t1 = wmma16(kc1, bq[qt][1], t1);

      // Online softmax: lane (lo,q) holds keys j+0..7 / j+16..23; partner lane
      // holds j+8..15 / j+24..31. One xor-16 completes the 32-key reduction.
      float mx = -3.4e38f;
#pragma unroll
      for (int r = 0; r < 8; r++) { mx = fmaxf(mx, t0[r]); mx = fmaxf(mx, t1[r]); }
      mx = fmaxf(mx, __shfl_xor(mx, 16, 32));
      float m_new = fmaxf(m_run[qt], mx);
      float alpha = __expf(m_run[qt] - m_new);

      float p0[8], p1[8], s = 0.f;
#pragma unroll
      for (int r = 0; r < 8; r++) {
        p0[r] = __expf(t0[r] - m_new);
        p1[r] = __expf(t1[r] - m_new);
        s += p0[r] + p1[r];
      }
      s += __shfl_xor(s, 16, 32);
      l_run[qt] = l_run[qt] * alpha + s;
      m_run[qt] = m_new;

      // P^T in B-layout (32 keys x 16 q):
      //  lanes 0-15:  e[0..7]=own t0,     e[8..15]=partner t0
      //  lanes 16-31: e[0..7]=partner t1, e[8..15]=own t1
#pragma unroll
      for (int r = 0; r < 8; r++) {
        float q0 = __shfl_xor(p0[r], 16, 32);
        float q1 = __shfl_xor(p1[r], 16, 32);
        pb[qt].e[r]     = (half_t)(hi ? q1 : p0[r]);
        pb[qt].e[r + 8] = (half_t)(hi ? p1[r] : q0);
      }

#pragma unroll
      for (int t = 0; t < 4; t++)
#pragma unroll
        for (int r = 0; r < 8; r++) acc[qt][t][r] *= alpha;
    }

    // O^T += V^T_tile * P^T : stream V A-tiles (16 d x 32 keys) from LDS.
#pragma unroll
    for (int t = 0; t < 4; t++) {
      v16h va = load_A(vb, t * 16 + laneLo, 32, 0, hi);
      acc[0][t] = wmma16(va, pb[0].v, acc[0][t]);
      acc[1][t] = wmma16(va, pb[1].v, acc[1][t]);
    }

    if (more) {                                      // publish next chunk
      *(uint4*)(&Kbuf[p ^ 1][krow * 64 + kcol]) = kreg;
      *(uint4*)(&Vbuf[p ^ 1][vrow * 32 + vcol]) = vreg;
    }
  }

  // Epilogue: O[b, n, h, d] f16; lane (hi,q) owns d = t*16 + hi*8 + 0..7.
  const int b_ = bh >> 4, h = bh & 15;
#pragma unroll
  for (int qt = 0; qt < 2; qt++) {
    float inv_l = 1.f / l_run[qt];
    size_t rowoff = ((size_t)(b_ * 2048 + qbase + qt * 16 + laneLo)) * 1024 + h * 64 + hi * 8;
#pragma unroll
    for (int t = 0; t < 4; t++) {
      union { half_t h8[8]; uint4 u; } r;
#pragma unroll
      for (int r8 = 0; r8 < 8; r8++) r.h8[r8] = (half_t)(acc[qt][t][r8] * inv_l);
      *(uint4*)(O + rowoff + t * 16) = r.u;
    }
  }
}

// ---------------- Output proj GEMM: [8192,1024] x [1024,1024] + bias -> fp32 ----------------
__global__ __launch_bounds__(256) void k_gemm_proj(const half_t* __restrict__ A, const half_t* __restrict__ Wt,
                                                   const float* __restrict__ bias, float* __restrict__ out) {
  const int lane = threadIdx.x & 31, wave = threadIdx.x >> 5;
  const int laneLo = lane & 15, hi = lane >> 4;
  const int Mbase = blockIdx.x * 128 + (wave & 1) * 64;
  const int Nbase = blockIdx.y * 128 + (wave >> 1) * 32;

  v8f acc[4][2] = {};
  for (int k = 0; k < 1024; k += 32) {
    v16h a0 = load_A(A, Mbase +  0 + laneLo, 1024, k, hi);
    v16h a1 = load_A(A, Mbase + 16 + laneLo, 1024, k, hi);
    v16h a2 = load_A(A, Mbase + 32 + laneLo, 1024, k, hi);
    v16h a3 = load_A(A, Mbase + 48 + laneLo, 1024, k, hi);
    v16h b0 = load_B(Wt, Nbase +  0 + laneLo, 1024, k, hi);
    v16h b1 = load_B(Wt, Nbase + 16 + laneLo, 1024, k, hi);
    acc[0][0] = wmma16(a0, b0, acc[0][0]);  acc[0][1] = wmma16(a0, b1, acc[0][1]);
    acc[1][0] = wmma16(a1, b0, acc[1][0]);  acc[1][1] = wmma16(a1, b1, acc[1][1]);
    acc[2][0] = wmma16(a2, b0, acc[2][0]);  acc[2][1] = wmma16(a2, b1, acc[2][1]);
    acc[3][0] = wmma16(a3, b0, acc[3][0]);  acc[3][1] = wmma16(a3, b1, acc[3][1]);
  }

#pragma unroll
  for (int nt = 0; nt < 2; nt++) {
    int c = Nbase + nt * 16 + laneLo;
    float bv = bias[c];
#pragma unroll
    for (int mt = 0; mt < 4; mt++) {
      int row0 = Mbase + mt * 16 + hi * 8;
#pragma unroll
      for (int r8 = 0; r8 < 8; r8++)
        out[(size_t)(row0 + r8) * 1024 + c] = acc[mt][nt][r8] + bv;
    }
  }
}

extern "C" void kernel_launch(void* const* d_in, const int* in_sizes, int n_in,
                              void* d_out, int out_size, void* d_ws, size_t ws_size,
                              hipStream_t stream) {
  const float* x      = (const float*)d_in[0];  // [4,2048,1024]
  const float* w_qkv  = (const float*)d_in[1];  // [1024,3072]
  const float* b_qkv  = (const float*)d_in[2];  // [3072]
  const float* w_proj = (const float*)d_in[3];  // [1024,1024]
  const float* b_proj = (const float*)d_in[4];  // [1024]
  float* out = (float*)d_out;                   // [4,2048,1024] fp32

  // Workspace layout (f16 elements): ~92 MB total
  half_t* xb  = (half_t*)d_ws;                          // 8192*1024
  half_t* wqt = xb  + (size_t)8192 * 1024;              // 3072*1024 (w_qkv^T)
  half_t* wpt = wqt + (size_t)3072 * 1024;              // 1024*1024 (w_proj^T)
  half_t* Qb  = wpt + (size_t)1024 * 1024;              // [B,H,N,64]
  half_t* Kb  = Qb  + (size_t)8388608;                  // [B,H,N,64]
  half_t* Vtb = Kb  + (size_t)8388608;                  // [B,H,64,N]
  half_t* AOb = Vtb + (size_t)8388608;                  // [B,N,H,64]

  k_cvt<<<(8192 * 1024 / 4 + 255) / 256, 256, 0, stream>>>(x, xb, 8192 * 1024);
  k_transpose<<<dim3(3072 / 32, 1024 / 32), 256, 0, stream>>>(w_qkv, wqt, 1024, 3072);
  k_transpose<<<dim3(1024 / 32, 1024 / 32), 256, 0, stream>>>(w_proj, wpt, 1024, 1024);
  k_gemm_qkv<<<dim3(64, 24), 256, 0, stream>>>(xb, wqt, b_qkv, Qb, Kb, Vtb);
  k_attn<<<dim3(64, 8), 256, 0, stream>>>(Qb, Kb, Vtb, AOb);
  k_gemm_proj<<<dim3(64, 8), 256, 0, stream>>>(AOb, wpt, b_proj, out);
}
